// HybridAttention_1254130450885
// MI455X (gfx1250) — matrix operand order repaired and checked by reading, not compiled, not run on hardware
//
#include <hip/hip_runtime.h>
#include <hip/hip_bf16.h>

// ---------------------------------------------------------------------------
// HybridAttention for MI455X (gfx1250): bf16 WMMA pipeline
//   k_cvt   : fp32 -> bf16 for x and weights
//   k_qkv   : Q/K/V projections (V stored transposed [b,h,hd,t]), 32x64/wave
//   k_attn  : windowed+global attention, WMMA for QK^T and P*V
//   k_oproj : output projection (32x64/wave), f32 result
// ---------------------------------------------------------------------------

typedef __bf16 bf16;
typedef __attribute__((ext_vector_type(16))) __bf16 v16bf;
typedef __attribute__((ext_vector_type(8)))  __bf16 v8bf;
typedef __attribute__((ext_vector_type(4)))  __bf16 v4bf;
typedef __attribute__((ext_vector_type(8)))  float  v8f;

union Frag16 { v16bf v; v8bf h[2]; };

#define B_   2
#define T_   8192
#define D_   1024
#define H_   16
#define HD_  64
#define WIN_ 128
#define G_   64
#define NW_  (T_ / WIN_)        // 64
#define BH_  (B_ * H_)          // 32
#define M_   (B_ * T_)          // 16384

__device__ __forceinline__ v8bf load8(const bf16* p) {
  return *reinterpret_cast<const v8bf*>(p);
}

__device__ __forceinline__ v8f wmma_bf16(v16bf a, v16bf b, v8f c) {
  // D = A(16x32 bf16) * B(32x16 bf16) + C(16x16 f32)
  return __builtin_amdgcn_wmma_f32_16x16x32_bf16(
      /*neg_a=*/false, a, /*neg_b=*/false, b,
      /*c_mod=*/(short)0, c, /*reuse_a=*/false, /*reuse_b=*/false);
}

// ------------------------------- fp32 -> bf16 ------------------------------
__global__ void k_cvt(const float4* __restrict__ in, v4bf* __restrict__ out, int n4) {
  int i = blockIdx.x * blockDim.x + threadIdx.x;
  if (i < n4) {
    float4 f = in[i];
    v4bf o;
    o[0] = (bf16)f.x; o[1] = (bf16)f.y; o[2] = (bf16)f.z; o[3] = (bf16)f.w;
    out[i] = o;
  }
}

// ------------------------- QKV projection (x @ W^T) ------------------------
// grid = (M/256, D/64, 3); block = 256 (8 waves). Wave: 32x64 output tile
// (2 M-subtiles x 4 N-subtiles) -> each B-fragment used by 2 WMMAs, each
// A-fragment by 4. A = x[m,k] (K-contiguous), B[k,n] = W[n,k] (K-contiguous).
__global__ __launch_bounds__(256)
void k_qkv(const bf16* __restrict__ X,
           const bf16* __restrict__ Wq, const bf16* __restrict__ Wk,
           const bf16* __restrict__ Wv,
           bf16* __restrict__ Qb, bf16* __restrict__ Kb, bf16* __restrict__ Vt) {
  const int lane = threadIdx.x & 31;
  const int wave = threadIdx.x >> 5;
  const int ln = lane & 15;   // column / row-within-tile index
  const int lg = lane >> 4;   // K-chunk group (0/1)
  const int row0 = blockIdx.x * 256 + wave * 32;
  const int col0 = blockIdx.y * 64;
  const int sel  = blockIdx.z;                 // 0=Q 1=K 2=V
  const bf16* Wm = (sel == 0) ? Wq : ((sel == 1) ? Wk : Wv);

  const bf16* arow0 = X + (size_t)(row0 + ln) * D_;
  const bf16* arow1 = X + (size_t)(row0 + 16 + ln) * D_;
  v8f acc[8];
#pragma unroll
  for (int j = 0; j < 8; ++j) acc[j] = (v8f){0,0,0,0,0,0,0,0};

  for (int k0 = 0; k0 < D_; k0 += 32) {
    Frag16 a0, a1;
    a0.h[0] = load8(arow0 + k0 + lg * 8);        // K = k0 + [0..7]/[8..15]
    a0.h[1] = load8(arow0 + k0 + 16 + lg * 8);   // K = k0+16 + ...
    a1.h[0] = load8(arow1 + k0 + lg * 8);
    a1.h[1] = load8(arow1 + k0 + 16 + lg * 8);
#pragma unroll
    for (int j = 0; j < 4; ++j) {
      const bf16* brow = Wm + (size_t)(col0 + j * 16 + ln) * D_ + k0 + lg * 8;
      Frag16 b;
      b.h[0] = load8(brow);
      b.h[1] = load8(brow + 16);
      acc[j]     = wmma_bf16(a0.v, b.v, acc[j]);
      acc[4 + j] = wmma_bf16(a1.v, b.v, acc[4 + j]);
    }
  }

  // C layout: VGPR r -> row (r + 8*lg), column ln within tile
#pragma unroll
  for (int mt = 0; mt < 2; ++mt) {
    const int rbase = row0 + mt * 16;
    if (sel == 2) {
#pragma unroll
      for (int j = 0; j < 4; ++j)
#pragma unroll
        for (int r = 0; r < 8; ++r) {
          int m = rbase + r + lg * 8;
          int n = col0 + j * 16 + ln;
          int bb = m >> 13, t = m & (T_ - 1);
          int hh = n >> 6,  d = n & (HD_ - 1);
          size_t bh = (size_t)(bb * H_ + hh);
          Vt[(bh * HD_ + d) * T_ + t] = (bf16)acc[mt * 4 + j][r];  // [b,h,hd,t]
        }
    } else {
      bf16* dst = sel ? Kb : Qb;
#pragma unroll
      for (int j = 0; j < 4; ++j)
#pragma unroll
        for (int r = 0; r < 8; ++r) {
          int m = rbase + r + lg * 8;
          int n = col0 + j * 16 + ln;
          int bb = m >> 13, t = m & (T_ - 1);
          int hh = n >> 6,  d = n & (HD_ - 1);
          size_t bh = (size_t)(bb * H_ + hh);
          dst[(bh * T_ + t) * HD_ + d] = (bf16)acc[mt * 4 + j][r];  // [b,h,t,hd]
        }
    }
  }
}

// ------------------------------- attention ---------------------------------
// grid = (NW, B*H); block = 256 (8 waves), wave handles 16 query rows.
// Keys: columns 0..63 = global (t=0..63), 64..191 = local window.
#define ATT_PITCH 200   // bf16 elements per LDS row: 192 + 8 pad (bank spread)

__global__ __launch_bounds__(256)
void k_attn(const bf16* __restrict__ Qb, const bf16* __restrict__ Kb,
            const bf16* __restrict__ Vt, bf16* __restrict__ Ob) {
  __shared__ bf16 lds[8 * 16 * ATT_PITCH];   // 51200 B: P staging per wave

  const int lane = threadIdx.x & 31;
  const int wave = threadIdx.x >> 5;
  const int ln = lane & 15;
  const int lg = lane >> 4;
  const int win = blockIdx.x;
  const int bh  = blockIdx.y;
  const size_t baseQK = (size_t)bh * T_ * HD_;
  const int q0 = win * WIN_ + wave * 16;

  const bf16* qrow = Qb + baseQK + (size_t)(q0 + ln) * HD_;

  // ---- S = Q K^T : 12 chunks of 16 keys, hd split into 2 WMMA K-steps
  v8f s[12];
#pragma unroll
  for (int c = 0; c < 12; ++c) s[c] = (v8f){0,0,0,0,0,0,0,0};

#pragma unroll
  for (int ks = 0; ks < 2; ++ks) {
    const int k0 = ks * 32;
    Frag16 a;
    a.h[0] = load8(qrow + k0 + lg * 8);
    a.h[1] = load8(qrow + k0 + 16 + lg * 8);
#pragma unroll
    for (int c = 0; c < 12; ++c) {
      const int tkey = (c < 4) ? (c * 16 + ln) : (win * WIN_ + (c - 4) * 16 + ln);
      const bf16* krow = Kb + baseQK + (size_t)tkey * HD_ + k0 + lg * 8;
      Frag16 b;
      b.h[0] = load8(krow);
      b.h[1] = load8(krow + 16);
      s[c] = wmma_bf16(a.v, b.v, s[c]);
    }
  }

  // ---- softmax over 192 columns per row (row = r + 8*lg, cols across lanes)
  const float sc_l2e = 0.125f * 1.44269504088896f;   // 1/sqrt(64) * log2(e)
  float inv_sum[8], mx[8];
#pragma unroll
  for (int r = 0; r < 8; ++r) {
    float v = s[0][r];
#pragma unroll
    for (int c = 1; c < 12; ++c) v = fmaxf(v, s[c][r]);
#pragma unroll
    for (int off = 8; off >= 1; off >>= 1)      // reduce within 16-lane half
      v = fmaxf(v, __shfl_xor(v, off, 32));
    mx[r] = v;
  }
#pragma unroll
  for (int r = 0; r < 8; ++r) {
    float sum = 0.f;
#pragma unroll
    for (int c = 0; c < 12; ++c) {
      float p = __builtin_amdgcn_exp2f((s[c][r] - mx[r]) * sc_l2e);
      s[c][r] = p;
      sum += p;
    }
#pragma unroll
    for (int off = 8; off >= 1; off >>= 1)
      sum += __shfl_xor(sum, off, 32);
    inv_sum[r] = __builtin_amdgcn_rcpf(sum);
  }

  // ---- stage normalized P (bf16) in LDS to re-form A-fragment layout
  bf16* prow = lds + wave * 16 * ATT_PITCH;
#pragma unroll
  for (int c = 0; c < 12; ++c)
#pragma unroll
    for (int r = 0; r < 8; ++r)
      prow[(r + lg * 8) * ATT_PITCH + c * 16 + ln] = (bf16)(s[c][r] * inv_sum[r]);

  __syncthreads();   // defensive; each wave only touches its own region

  // ---- O = P (16x192) @ V (192x64): 6 K-steps, 4 N-tiles
  v8f o[4];
#pragma unroll
  for (int j = 0; j < 4; ++j) o[j] = (v8f){0,0,0,0,0,0,0,0};

#pragma unroll
  for (int ks = 0; ks < 6; ++ks) {
    const int k0 = ks * 32;
    Frag16 a;
    const bf16* pl = prow + ln * ATT_PITCH + k0;
    a.h[0] = load8(pl + lg * 16);
    a.h[1] = load8(pl + lg * 16 + 8);
    const int kk = k0 + lg * 16;                       // key index of 16-chunk
    const int tb = (kk < G_) ? kk : (win * WIN_ + kk - G_);
#pragma unroll
    for (int j = 0; j < 4; ++j) {
      const bf16* vrow = Vt + ((size_t)bh * HD_ + j * 16 + ln) * T_ + tb;
      Frag16 b;
      b.h[0] = load8(vrow);
      b.h[1] = load8(vrow + 8);
      o[j] = wmma_bf16(a.v, b.v, o[j]);
    }
  }

  // ---- store O as [b, t, h*64+d] so final projection is a plain GEMM
  const int bb = bh >> 4, hh = bh & 15;
#pragma unroll
  for (int j = 0; j < 4; ++j)
#pragma unroll
    for (int r = 0; r < 8; ++r) {
      const int t = q0 + r + lg * 8;
      const int d = j * 16 + ln;
      Ob[((size_t)bb * T_ + t) * D_ + hh * HD_ + d] = (bf16)o[j][r];
    }
}

// --------------------------- output projection -----------------------------
// grid = (M/256, D/64); wave computes 32x64 (same tiling as k_qkv), f32 out.
__global__ __launch_bounds__(256)
void k_oproj(const bf16* __restrict__ Ob, const bf16* __restrict__ Wo,
             float* __restrict__ out) {
  const int lane = threadIdx.x & 31;
  const int wave = threadIdx.x >> 5;
  const int ln = lane & 15;
  const int lg = lane >> 4;
  const int row0 = blockIdx.x * 256 + wave * 32;
  const int col0 = blockIdx.y * 64;

  const bf16* arow0 = Ob + (size_t)(row0 + ln) * D_;
  const bf16* arow1 = Ob + (size_t)(row0 + 16 + ln) * D_;
  v8f acc[8];
#pragma unroll
  for (int j = 0; j < 8; ++j) acc[j] = (v8f){0,0,0,0,0,0,0,0};

  for (int k0 = 0; k0 < D_; k0 += 32) {
    Frag16 a0, a1;
    a0.h[0] = load8(arow0 + k0 + lg * 8);
    a0.h[1] = load8(arow0 + k0 + 16 + lg * 8);
    a1.h[0] = load8(arow1 + k0 + lg * 8);
    a1.h[1] = load8(arow1 + k0 + 16 + lg * 8);
#pragma unroll
    for (int j = 0; j < 4; ++j) {
      const bf16* brow = Wo + (size_t)(col0 + j * 16 + ln) * D_ + k0 + lg * 8;
      Frag16 b;
      b.h[0] = load8(brow);
      b.h[1] = load8(brow + 16);
      acc[j]     = wmma_bf16(a0.v, b.v, acc[j]);
      acc[4 + j] = wmma_bf16(a1.v, b.v, acc[4 + j]);
    }
  }
#pragma unroll
  for (int mt = 0; mt < 2; ++mt)
#pragma unroll
    for (int j = 0; j < 4; ++j)
#pragma unroll
      for (int r = 0; r < 8; ++r)
        out[(size_t)(row0 + mt * 16 + r + lg * 8) * D_ + col0 + j * 16 + ln] =
            acc[mt * 4 + j][r];
}

// ------------------------------- launcher ----------------------------------
extern "C" void kernel_launch(void* const* d_in, const int* in_sizes, int n_in,
                              void* d_out, int out_size, void* d_ws, size_t ws_size,
                              hipStream_t stream) {
  (void)in_sizes; (void)n_in; (void)out_size; (void)ws_size;
  const float* x  = (const float*)d_in[0];
  const float* Wq = (const float*)d_in[1];
  const float* Wk = (const float*)d_in[2];
  const float* Wv = (const float*)d_in[3];
  const float* Wo = (const float*)d_in[4];

  const size_t SZ_X   = (size_t)M_ * D_;        // 16.7M elems
  const size_t SZ_W   = (size_t)D_ * D_;        // 1M elems
  const size_t SZ_QKV = (size_t)BH_ * T_ * HD_; // 16.7M elems

  bf16* p   = (bf16*)d_ws;
  bf16* xb  = p;  p += SZ_X;
  bf16* wqb = p;  p += SZ_W;
  bf16* wkb = p;  p += SZ_W;
  bf16* wvb = p;  p += SZ_W;
  bf16* wob = p;  p += SZ_W;
  bf16* Qb  = p;  p += SZ_QKV;
  bf16* Kb  = p;  p += SZ_QKV;
  bf16* Vt  = p;  p += SZ_QKV;
  bf16* Ob  = p;  p += SZ_QKV;

  // fp32 -> bf16
  {
    int n4 = (int)(SZ_X / 4);
    k_cvt<<<dim3((n4 + 255) / 256), 256, 0, stream>>>((const float4*)x, (v4bf*)xb, n4);
    int w4 = (int)(SZ_W / 4);
    k_cvt<<<dim3((w4 + 255) / 256), 256, 0, stream>>>((const float4*)Wq, (v4bf*)wqb, w4);
    k_cvt<<<dim3((w4 + 255) / 256), 256, 0, stream>>>((const float4*)Wk, (v4bf*)wkb, w4);
    k_cvt<<<dim3((w4 + 255) / 256), 256, 0, stream>>>((const float4*)Wv, (v4bf*)wvb, w4);
    k_cvt<<<dim3((w4 + 255) / 256), 256, 0, stream>>>((const float4*)Wo, (v4bf*)wob, w4);
  }

  // Q/K/V projections
  k_qkv<<<dim3(M_ / 256, D_ / 64, 3), 256, 0, stream>>>(xb, wqb, wkb, wvb, Qb, Kb, Vt);

  // attention per (window, b*h)
  k_attn<<<dim3(NW_, BH_), 256, 0, stream>>>(Qb, Kb, Vt, Ob);

  // output projection -> f32
  k_oproj<<<dim3(M_ / 256, D_ / 64), 256, 0, stream>>>(Ob, wob, (float*)d_out);
}
